// DigitCapsules_74448963109435
// MI455X (gfx1250) — compile-verified, matrix operand order checked by reading, not run on hardware
//
#include <hip/hip_runtime.h>
#include <hip/hip_fp16.h>

// Problem dims (from reference): B=256, R=1152, C=10, IC=8, OC=16
#define B_      256
#define R_      1152
#define C_      10
#define IC_     8
#define OC_     16
#define RCHUNKS 32
#define RPER    (R_ / RCHUNKS)      // 36 routes per wave
#define SELEMS  (B_ * C_ * OC_)     // 40960 elements in s / vsum / v

typedef __attribute__((ext_vector_type(2)))  float    v2f;
typedef __attribute__((ext_vector_type(8)))  float    v8f;
typedef __attribute__((ext_vector_type(16))) _Float16 v16h;

// ---------------------------------------------------------------------------
// zero vsum at start of every launch
__global__ void caps_zero(float* __restrict__ vsum) {
    int i = blockIdx.x * blockDim.x + threadIdx.x;
    if (i < SELEMS) vsum[i] = 0.0f;
}

// ---------------------------------------------------------------------------
// Fused routing pass: one wave owns (b_tile of 16 rows, chunk of 36 routes).
// For each r: u[16b x 160co] via WMMA (10 tiles of 16x16, K=8),
// logits L[b,c] = sum_o u*vsum (butterfly reduce over o in 16-lane groups),
// softmax over c (per-lane, values replicated in group), accumulate c*u.
// Partial s written (non-atomically, deterministic) to spart[chunk].
__global__ __launch_bounds__(128) void caps_route(
    const float* __restrict__ x,      // [B,R,IC]
    const float* __restrict__ w,      // [R,C,IC,OC]
    const float* __restrict__ vsum,   // [B,C,OC]  (v1+...+v_{k-1})
    float*       __restrict__ spart)  // [RCHUNKS,B,C,OC]
{
    const int lane  = threadIdx.x & 31;
    const int wid   = blockIdx.x * (blockDim.x >> 5) + (threadIdx.x >> 5); // 0..511
    const int bt    = wid >> 5;          // 0..15 : b-tile
    const int ck    = wid & 31;          // 0..31 : r-chunk
    const int bbase = bt * 16;
    const int m     = lane & 15;         // row (A) / col o (B,D)
    const int half  = lane >> 4;         // 0: lanes 0-15, 1: lanes 16-31

    // Preload vsum for this b-tile: vs[j][c] = vsum[bbase+j+8*half, c, o=m]
    float vs[8][C_];
#pragma unroll
    for (int j = 0; j < 8; ++j) {
        const int b = bbase + j + 8 * half;
#pragma unroll
        for (int c = 0; c < C_; ++c)
            vs[j][c] = vsum[(b * C_ + c) * OC_ + m];
    }

    const v8f vzero = {};
    v8f sacc[C_];
#pragma unroll
    for (int c = 0; c < C_; ++c) sacc[c] = vzero;

    const int r0 = ck * RPER;
    for (int r = r0; r < r0 + RPER; ++r) {
        v8f u[C_];

#if __has_builtin(__builtin_amdgcn_wmma_f32_16x16x4_f32)
        // ---- FP32 WMMA path: K=8 as two 16x16x4 steps --------------------
        // A 16x4 f32 layout: lanes 0-15 -> (K0,K1), lanes 16-31 -> (K2,K3)
        const float* xp = x + ((size_t)(bbase + m) * R_ + r) * IC_ + half * 2;
        v2f a0; a0.x = xp[0]; a0.y = xp[1];   // K = half*2, half*2+1
        v2f a1; a1.x = xp[4]; a1.y = xp[5];   // K = 4 + half*2, +1
#pragma unroll
        for (int c = 0; c < C_; ++c) {
            // B 4x16 f32 layout: lane n<16 -> rows (K0,K1); n>=16 -> (K2,K3)
            const float* wp = w + (((size_t)r * C_ + c) * IC_ + half * 2) * OC_ + m;
            v2f b0; b0.x = wp[0];  b0.y = wp[16];   // i = half*2, half*2+1
            v2f b1; b1.x = wp[64]; b1.y = wp[80];   // i = 4+half*2, +1
            v8f acc = vzero;
            acc = __builtin_amdgcn_wmma_f32_16x16x4_f32(
                      false, a0, false, b0, (short)0, acc, false, false);
            acc = __builtin_amdgcn_wmma_f32_16x16x4_f32(
                      false, a1, false, b1, (short)0, acc, false, false);
            u[c] = acc;
        }
#else
        // ---- fallback: f16 WMMA 16x16x32, K padded 8 -> 32 ---------------
        // A 16-bit 16x32: lanes 0-15 elems 0-7 = K0..7; everything else 0.
        const float hsel = (half == 0) ? 1.0f : 0.0f;
        const float* xp = x + ((size_t)(bbase + m) * R_ + r) * IC_;
        v16h a = {};
#pragma unroll
        for (int i = 0; i < IC_; ++i) a[i] = (_Float16)(xp[i] * hsel);
#pragma unroll
        for (int c = 0; c < C_; ++c) {
            // B 16-bit 32x16: lanes 0-15 elems 0-7 = rows K0..7, col o=m.
            const float* wp = w + (((size_t)r * C_ + c) * IC_) * OC_ + m;
            v16h bmat = {};
#pragma unroll
            for (int i = 0; i < IC_; ++i)
                bmat[i] = (_Float16)(wp[i * OC_] * hsel);
            v8f acc = vzero;
            acc = __builtin_amdgcn_wmma_f32_16x16x32_f16(
                      false, a, false, bmat, (short)0, acc, false, false);
            u[c] = acc;
        }
#endif

        // ---- logits, softmax over capsules, accumulate s -----------------
#pragma unroll
        for (int j = 0; j < 8; ++j) {
            float L[C_];
#pragma unroll
            for (int c = 0; c < C_; ++c) {
                float p = u[c][j] * vs[j][c];
                p += __shfl_xor(p, 1, 32);
                p += __shfl_xor(p, 2, 32);
                p += __shfl_xor(p, 4, 32);
                p += __shfl_xor(p, 8, 32);   // sum over o within 16-lane group
                L[c] = p;
            }
            float mx = L[0];
#pragma unroll
            for (int c = 1; c < C_; ++c) mx = fmaxf(mx, L[c]);
            float den = 0.0f;
#pragma unroll
            for (int c = 0; c < C_; ++c) { L[c] = __expf(L[c] - mx); den += L[c]; }
            const float inv = 1.0f / den;
#pragma unroll
            for (int c = 0; c < C_; ++c)
                sacc[c][j] += (L[c] * inv) * u[c][j];
        }
    }

    // write this chunk's partial s (unique region -> plain stores)
    float* sp = spart + (size_t)ck * SELEMS;
#pragma unroll
    for (int j = 0; j < 8; ++j) {
        const int b = bbase + j + 8 * half;
#pragma unroll
        for (int c = 0; c < C_; ++c)
            sp[(b * C_ + c) * OC_ + m] = sacc[c][j];
    }
}

// ---------------------------------------------------------------------------
// Reduce chunk partials (fixed order -> deterministic), squash, vsum += v.
// Thread idx = (b*C + c)*16 + o, so 16 consecutive lanes share (b,c).
__global__ void caps_squash(const float* __restrict__ spart,
                            float* __restrict__ vsum,
                            float* __restrict__ out,
                            int write_out)
{
    const int idx = blockIdx.x * blockDim.x + threadIdx.x;
    if (idx >= SELEMS) return;
    float s = 0.0f;
    for (int ck = 0; ck < RCHUNKS; ++ck)
        s += spart[(size_t)ck * SELEMS + idx];

    float n2 = s * s;
    n2 += __shfl_xor(n2, 1, 32);
    n2 += __shfl_xor(n2, 2, 32);
    n2 += __shfl_xor(n2, 4, 32);
    n2 += __shfl_xor(n2, 8, 32);            // ||s||^2 over o (16-lane group)
    const float norm = sqrtf(n2);
    const float v = s * (n2 / (1.0f + n2)) / (norm + 1e-8f);
    vsum[idx] += v;
    if (write_out) out[idx] = v;
}

// ---------------------------------------------------------------------------
extern "C" void kernel_launch(void* const* d_in, const int* in_sizes, int n_in,
                              void* d_out, int out_size, void* d_ws, size_t ws_size,
                              hipStream_t stream) {
    const float* x = (const float*)d_in[0];   // [B,R,IC]
    const float* w = (const float*)d_in[1];   // [1,R,C,IC,OC]
    float* out = (float*)d_out;               // [B,C,OC]

    float* spart = (float*)d_ws;                              // RCHUNKS*SELEMS
    float* vsum  = spart + (size_t)RCHUNKS * SELEMS;          // SELEMS
    if (ws_size < (size_t)(RCHUNKS + 1) * SELEMS * sizeof(float)) return;

    caps_zero<<<(SELEMS + 255) / 256, 256, 0, stream>>>(vsum);
    for (int it = 0; it < 3; ++it) {
        // 512 waves: 16 b-tiles x 32 r-chunks; 4 waves / block
        caps_route<<<128, 128, 0, stream>>>(x, w, vsum, spart);
        caps_squash<<<(SELEMS + 255) / 256, 256, 0, stream>>>(
            spart, vsum, out, it == 2 ? 1 : 0);
    }
}
